// CausalSelfAttentionGQA_83863531422015
// MI455X (gfx1250) — matrix operand order repaired and checked by reading, not compiled, not run on hardware
//
#include <hip/hip_runtime.h>
#include <hip/hip_bf16.h>
#include <math.h>

// ---------------- types ----------------
typedef __attribute__((ext_vector_type(16))) __bf16 v16bf;
typedef __attribute__((ext_vector_type(8)))  float  v8f;
typedef __attribute__((ext_vector_type(4)))  unsigned int u32x4;
typedef __attribute__((ext_vector_type(4)))  unsigned int v4u;
typedef __attribute__((ext_vector_type(4)))  int v4i;
typedef __attribute__((ext_vector_type(8)))  int v8i;

union Frag {
  v16bf v;
  u32x4 u[2];
};

__device__ __forceinline__ unsigned short f2bf(float f) {
  union { float f; unsigned int u; } x; x.f = f;
  unsigned int r = x.u + 0x7FFFu + ((x.u >> 16) & 1u);
  return (unsigned short)(r >> 16);
}

// --------- Tensor Data Mover (TDM) path: gate per-pass / per-toolchain ------
#if defined(__AMDGCN__) && defined(__has_builtin)
#if __has_builtin(__builtin_amdgcn_tensor_load_to_lds) && __has_builtin(__builtin_amdgcn_s_wait_tensorcnt)
#define USE_TDM 1
#endif
#endif
#ifndef USE_TDM
#define USE_TDM 0
#endif

#if USE_TDM
// 2-D tile DMA: global (row-major, bf16, row stride = stride0 elements)
//   -> LDS packed tile [tile_d1][tile_d0] bf16.
// Descriptor bit layout per CDNA5 ISA D# group0/group1 tables.
__device__ __forceinline__ void tdm_load_2d(unsigned lds_addr, const void* gptr,
                                            unsigned tensor_d0, unsigned tensor_d1,
                                            unsigned tile_d0, unsigned tile_d1,
                                            unsigned long long stride0) {
  unsigned long long ga = (unsigned long long)(size_t)gptr;
  v4u g0;
  g0[0] = 1u;                                   // count=1, user descriptor
  g0[1] = lds_addr;                             // [63:32] lds_addr (bytes)
  g0[2] = (unsigned)(ga & 0xffffffffu);         // [95:64] global_addr lo
  g0[3] = (unsigned)((ga >> 32) & 0x01ffffffu)  // [120:96] global_addr hi
        | (2u << 30);                           // [127:126] type=2 (image)
  v8i g1;
  g1[0] = (int)(1u << 16);                                        // data_size=1 (2B), mask=0
  g1[1] = (int)((tensor_d0 & 0xffffu) << 16);                     // [63:48] tensor_dim0 lo16
  g1[2] = (int)((tensor_d0 >> 16) | ((tensor_d1 & 0xffffu) << 16));
  g1[3] = (int)((tensor_d1 >> 16) | (tile_d0 << 16));             // [127:112] tile_dim0
  g1[4] = (int)(tile_d1 & 0xffffu);                               // tile_dim1, tile_dim2=0
  g1[5] = (int)(unsigned)(stride0 & 0xffffffffu);                 // tensor_dim0_stride lo
  g1[6] = (int)(unsigned)((stride0 >> 32) & 0xffffu);             // stride hi16, stride1=0
  g1[7] = 0;
  v4i z4 = {0, 0, 0, 0};
#if __clang_major__ >= 23
  v8i z8 = {0, 0, 0, 0, 0, 0, 0, 0};
  __builtin_amdgcn_tensor_load_to_lds(g0, g1, z4, z4, z8, 0);
#else
  __builtin_amdgcn_tensor_load_to_lds(g0, g1, z4, z4, 0);
#endif
}
#endif  // USE_TDM

// ---------------- conversion kernels ----------------
__global__ void k_cvt_bf16(const float* __restrict__ in,
                           unsigned short* __restrict__ out, int n) {
  int i = blockIdx.x * blockDim.x + threadIdx.x;
  if (i < n) out[i] = f2bf(in[i]);
}

// in: [R][C] fp32 row-major  ->  out: [C][R] bf16 (i.e. transposed)
__global__ void k_cvt_bf16_T(const float* __restrict__ in,
                             unsigned short* __restrict__ out, int R, int C) {
  int i = blockIdx.x * blockDim.x + threadIdx.x;
  if (i < R * C) {
    int r = i / C, c = i - r * C;
    out[(size_t)c * R + r] = f2bf(in[i]);
  }
}

// ---------------- bf16 GEMM: C[M][N] = A[M][K] * Bt[N][K]^T ----------------
// 256 threads = 8 waves; block computes 128x128 of C; wave computes 64x32.
// LDS tiles staged by the Tensor Data Mover (one wave issues 2 descriptors
// per K-step, waits TENSORcnt==0, workgroup barrier releases consumers).
__global__ __launch_bounds__(256) void k_gemm_bf16(
    const unsigned short* __restrict__ A,
    const unsigned short* __restrict__ Bt,
    float* __restrict__ C,
    int M, int N, int K) {
  __shared__ alignas(16) unsigned short lA[128][32];
  __shared__ alignas(16) unsigned short lB[128][32];
  const int tid  = threadIdx.x;
  const int lane = tid & 31;
  const int wave = tid >> 5;
  const int wm   = wave >> 2;   // 0..1
  const int wn   = wave & 3;    // 0..3
  const int half = lane >> 4;
  const int l16  = lane & 15;
  const int m0   = blockIdx.y * 128;
  const int n0   = blockIdx.x * 128;

  v8f acc[4][2];
#pragma unroll
  for (int i = 0; i < 4; ++i)
#pragma unroll
    for (int j = 0; j < 2; ++j)
#pragma unroll
      for (int r = 0; r < 8; ++r) acc[i][j][r] = 0.0f;

#if !USE_TDM
  const int lr = tid >> 1;          // 0..127 : row of tile
  const int lc = (tid & 1) * 16;    // 0 or 16 : col within 32-wide slice
#else
  const unsigned ldsA = (unsigned)(size_t)&lA[0][0];
  const unsigned ldsB = (unsigned)(size_t)&lB[0][0];
#endif

  for (int k0 = 0; k0 < K; k0 += 32) {
    __syncthreads();
#if USE_TDM
    if (tid < 32) {  // wave 0 drives the TDM
      tdm_load_2d(ldsA, A + (size_t)m0 * K + k0, (unsigned)K, (unsigned)M,
                  32u, 128u, (unsigned long long)K);
      tdm_load_2d(ldsB, Bt + (size_t)n0 * K + k0, (unsigned)K, (unsigned)N,
                  32u, 128u, (unsigned long long)K);
      __builtin_amdgcn_s_wait_tensorcnt(0);
    }
#else
    {
      const u32x4* sa = (const u32x4*)(A + (size_t)(m0 + lr) * K + k0 + lc);
      u32x4* da = (u32x4*)&lA[lr][lc];
      da[0] = sa[0]; da[1] = sa[1];
      const u32x4* sb = (const u32x4*)(Bt + (size_t)(n0 + lr) * K + k0 + lc);
      u32x4* db = (u32x4*)&lB[lr][lc];
      db[0] = sb[0]; db[1] = sb[1];
    }
#endif
    __syncthreads();

    Frag af[4], bfr[2];
#pragma unroll
    for (int mt = 0; mt < 4; ++mt) {
      int row = wm * 64 + mt * 16 + l16;
      // 16-bit A layout: lanes 0-15 hold K 0-7 / 16-23; lanes 16-31 hold K 8-15 / 24-31
      af[mt].u[0] = *(const u32x4*)&lA[row][half * 8];
      af[mt].u[1] = *(const u32x4*)&lA[row][16 + half * 8];
    }
#pragma unroll
    for (int nt = 0; nt < 2; ++nt) {
      int row = wn * 32 + nt * 16 + l16;
      // 16-bit B layout: lanes 0-15 hold K 0-7 / 16-23; lanes 16-31 hold K 8-15 / 24-31
      bfr[nt].u[0] = *(const u32x4*)&lB[row][half * 8];
      bfr[nt].u[1] = *(const u32x4*)&lB[row][16 + half * 8];
    }
#pragma unroll
    for (int mt = 0; mt < 4; ++mt)
#pragma unroll
      for (int nt = 0; nt < 2; ++nt)
        acc[mt][nt] = __builtin_amdgcn_wmma_f32_16x16x32_bf16(
            false, af[mt].v, false, bfr[nt].v, (short)0, acc[mt][nt], false, false);
  }

  // C layout: lanes 0-15 -> M = r, lanes 16-31 -> M = 8 + r; N = lane&15
#pragma unroll
  for (int mt = 0; mt < 4; ++mt)
#pragma unroll
    for (int nt = 0; nt < 2; ++nt) {
      float* cp = C + (size_t)(m0 + wm * 64 + mt * 16 + half * 8) * N
                    + (n0 + wn * 32 + nt * 16 + l16);
#pragma unroll
      for (int r = 0; r < 8; ++r) cp[(size_t)r * N] = acc[mt][nt][r];
    }
}

// ---------------- RoPE + reference-permutation scatter ----------------
// Reference reshape semantics: for Q, effective head h = t%16, position
// p = (kv*4+g)*256 + t/16  =>  t = (p%256)*16 + h, head_orig = p/256.
// For K/V: c = t%4, p = kv*1024 + t/4  =>  t = (p%1024)*4 + c, kv = p/1024.
// RoPE uses freqs_cis[p]. Outputs: Q[16][4096][128], K[4][4096][128],
// Vt[4][128][4096] (transposed for P*V B-fragments), all bf16.
__global__ void k_rope_scatter(const float* __restrict__ qkv,
                               const float* __restrict__ fc,
                               unsigned short* __restrict__ Qo,
                               unsigned short* __restrict__ Ko,
                               unsigned short* __restrict__ Vt) {
  const int QCNT = 16 * 4096 * 64;
  const int KCNT = 4 * 4096 * 64;
  int idx = blockIdx.x * blockDim.x + threadIdx.x;
  if (idx < QCNT) {
    int h   = idx / (4096 * 64);
    int rem = idx - h * (4096 * 64);
    int p   = rem >> 6;
    int m   = rem & 63;
    int d   = m * 2;
    int t   = (p & 255) * 16 + h;
    int ho  = p >> 8;
    int base = (ho >> 2) * 768 + (ho & 3) * 128;
    float a = qkv[(size_t)t * 3072 + base + d];
    float b = qkv[(size_t)t * 3072 + base + d + 1];
    float c = fc[((size_t)p * 64 + m) * 2 + 0];
    float s = fc[((size_t)p * 64 + m) * 2 + 1];
    size_t o = ((size_t)h * 4096 + p) * 128 + d;
    Qo[o]     = f2bf(a * c - b * s);
    Qo[o + 1] = f2bf(b * c + a * s);
  } else if (idx < QCNT + KCNT) {
    int i2  = idx - QCNT;
    int cH  = i2 / (4096 * 64);
    int rem = i2 - cH * (4096 * 64);
    int p   = rem >> 6;
    int m   = rem & 63;
    int d   = m * 2;
    int t   = (p & 1023) * 4 + cH;
    int kv  = p >> 10;
    int base = kv * 768 + 512;
    float a = qkv[(size_t)t * 3072 + base + d];
    float b = qkv[(size_t)t * 3072 + base + d + 1];
    float c = fc[((size_t)p * 64 + m) * 2 + 0];
    float s = fc[((size_t)p * 64 + m) * 2 + 1];
    size_t o = ((size_t)cH * 4096 + p) * 128 + d;
    Ko[o]     = f2bf(a * c - b * s);
    Ko[o + 1] = f2bf(b * c + a * s);
  } else if (idx < QCNT + 2 * KCNT) {
    int i2  = idx - QCNT - KCNT;
    int cH  = i2 / (4096 * 64);
    int rem = i2 - cH * (4096 * 64);
    int p   = rem >> 6;
    int m   = rem & 63;
    int d   = m * 2;
    int t   = (p & 1023) * 4 + cH;
    int kv  = p >> 10;
    int base = kv * 768 + 640;
    float a = qkv[(size_t)t * 3072 + base + d];
    float b = qkv[(size_t)t * 3072 + base + d + 1];
    size_t o = ((size_t)cH * 128 + d) * 4096 + p;
    Vt[o]        = f2bf(a);
    Vt[o + 4096] = f2bf(b);
  }
}

// ---------------- banded flash attention, one wave per (head, 16-q tile) ----
__global__ __launch_bounds__(32) void k_attn(
    const unsigned short* __restrict__ Q,   // [16][4096][128] bf16
    const unsigned short* __restrict__ Kb,  // [4][4096][128]  bf16
    const unsigned short* __restrict__ Vt,  // [4][128][4096]  bf16
    unsigned short* __restrict__ Y) {       // [4096][2048]    bf16
  __shared__ alignas(16) unsigned short sP[16][32];
  const int head = blockIdx.y;
  const int q0   = blockIdx.x * 16;
  const int kvh  = head >> 2;
  const int lane = threadIdx.x;
  const int half = lane >> 4;
  const int l16  = lane & 15;
  const float scale = 0.08838834764831845f;  // 1/sqrt(128)
  const float NEGBIG = -1e30f;

  const unsigned short* Qh = Q  + ((size_t)head * 4096 + q0) * 128;
  const unsigned short* Kh = Kb + (size_t)kvh * 4096 * 128;
  const unsigned short* Vh = Vt + (size_t)kvh * 128 * 4096;

  // Q A-fragments for the 4 K-steps (HD=128 = 4*32)
  Frag qf[4];
  {
    const unsigned short* qrow = Qh + (size_t)l16 * 128;
#pragma unroll
    for (int kk = 0; kk < 4; ++kk) {
      qf[kk].u[0] = *(const u32x4*)(qrow + kk * 32 + half * 8);
      qf[kk].u[1] = *(const u32x4*)(qrow + kk * 32 + 16 + half * 8);
    }
  }

  v8f acc[8];
#pragma unroll
  for (int nt = 0; nt < 8; ++nt)
#pragma unroll
    for (int r = 0; r < 8; ++r) acc[nt][r] = 0.0f;
  float rm[8], rs[8];
#pragma unroll
  for (int r = 0; r < 8; ++r) { rm[r] = NEGBIG; rs[r] = 0.0f; }

  int jstart = q0 - 1023; if (jstart < 0) jstart = 0; jstart &= ~31;
  int jend = q0 + 15 + 1023 + 1; if (jend > 4096) jend = 4096;
  jend = (jend + 31) & ~31; if (jend > 4096) jend = 4096;

  for (int j0 = jstart; j0 < jend; j0 += 32) {
    // ---- S = Q * K^T for a 16x32 chunk ----
    v8f s[2];
#pragma unroll
    for (int sub = 0; sub < 2; ++sub) {
#pragma unroll
      for (int r = 0; r < 8; ++r) s[sub][r] = 0.0f;
      const unsigned short* krow = Kh + (size_t)(j0 + sub * 16 + l16) * 128;
      __builtin_prefetch(krow + 32 * 128, 0, 1);  // next chunk's K rows
#pragma unroll
      for (int kk = 0; kk < 4; ++kk) {
        Frag b;
        b.u[0] = *(const u32x4*)(krow + kk * 32 + half * 8);
        b.u[1] = *(const u32x4*)(krow + kk * 32 + 16 + half * 8);
        s[sub] = __builtin_amdgcn_wmma_f32_16x16x32_bf16(
            false, qf[kk].v, false, b.v, (short)0, s[sub], false, false);
      }
    }

    // ---- band mask + online softmax ----
    float p[2][8], tmax[8];
#pragma unroll
    for (int r = 0; r < 8; ++r) {
      int i = q0 + half * 8 + r;
#pragma unroll
      for (int sub = 0; sub < 2; ++sub) {
        int j = j0 + sub * 16 + l16;
        int dd = i - j; dd = dd < 0 ? -dd : dd;
        p[sub][r] = (dd <= 1023) ? s[sub][r] * scale : NEGBIG;
      }
      tmax[r] = fmaxf(p[0][r], p[1][r]);
    }
#pragma unroll
    for (int msk = 8; msk >= 1; msk >>= 1)
#pragma unroll
      for (int r = 0; r < 8; ++r)
        tmax[r] = fmaxf(tmax[r], __shfl_xor(tmax[r], msk, 32));

    float tsum[8];
#pragma unroll
    for (int r = 0; r < 8; ++r) {
      float mnew = fmaxf(rm[r], tmax[r]);
      float corr = __expf(rm[r] - mnew);
      rm[r] = mnew;
      float e0 = __expf(p[0][r] - mnew);
      float e1 = __expf(p[1][r] - mnew);
      p[0][r] = e0; p[1][r] = e1;
      tsum[r] = e0 + e1;
      rs[r] *= corr;
#pragma unroll
      for (int nt = 0; nt < 8; ++nt) acc[nt][r] *= corr;
    }
#pragma unroll
    for (int msk = 8; msk >= 1; msk >>= 1)
#pragma unroll
      for (int r = 0; r < 8; ++r)
        tsum[r] += __shfl_xor(tsum[r], msk, 32);
#pragma unroll
    for (int r = 0; r < 8; ++r) rs[r] += tsum[r];

    // ---- relayout P (C-layout) -> A-layout via LDS ----
    __syncthreads();
#pragma unroll
    for (int r = 0; r < 8; ++r) {
      sP[half * 8 + r][l16]      = f2bf(p[0][r]);
      sP[half * 8 + r][16 + l16] = f2bf(p[1][r]);
    }
    __syncthreads();
    Frag pf;
    pf.u[0] = *(const u32x4*)&sP[l16][half * 8];
    pf.u[1] = *(const u32x4*)&sP[l16][16 + half * 8];

    // ---- acc += P * V ----
#pragma unroll
    for (int nt = 0; nt < 8; ++nt) {
      const unsigned short* vrow = Vh + (size_t)(nt * 16 + l16) * 4096 + j0 + half * 8;
      Frag b;
      b.u[0] = *(const u32x4*)(vrow);
      b.u[1] = *(const u32x4*)(vrow + 16);
      acc[nt] = __builtin_amdgcn_wmma_f32_16x16x32_bf16(
          false, pf.v, false, b.v, (short)0, acc[nt], false, false);
    }
  }

  // ---- normalize + store y (row p, col head*128 + d) ----
#pragma unroll
  for (int r = 0; r < 8; ++r) {
    float inv = 1.0f / rs[r];
    int row = q0 + half * 8 + r;
    unsigned short* yp = Y + (size_t)row * 2048 + head * 128 + l16;
#pragma unroll
    for (int nt = 0; nt < 8; ++nt)
      yp[nt * 16] = f2bf(acc[nt][r] * inv);
  }
}

// ---------------- host launcher ----------------
extern "C" void kernel_launch(void* const* d_in, const int* in_sizes, int n_in,
                              void* d_out, int out_size, void* d_ws, size_t ws_size,
                              hipStream_t stream) {
  (void)in_sizes; (void)n_in; (void)out_size; (void)ws_size;
  const float* x      = (const float*)d_in[0];
  const float* freqs  = (const float*)d_in[1];
  const float* W_attn = (const float*)d_in[2];
  const float* W_proj = (const float*)d_in[3];
  float* out = (float*)d_out;
  char* ws = (char*)d_ws;

  // workspace layout (bytes, 256-aligned)
  unsigned short* x_bf = (unsigned short*)(ws + 0);           // 16 MB
  unsigned short* wa_t = (unsigned short*)(ws + 16777216);    // 12 MB
  unsigned short* wp_t = (unsigned short*)(ws + 29360128);    //  8 MB
  float*          qkv  = (float*)         (ws + 37748736);    // 48 MB
  unsigned short* Qb   = (unsigned short*)(ws + 88080384);    // 16 MB
  unsigned short* Kb   = (unsigned short*)(ws + 104857600);   //  4 MB
  unsigned short* Vb   = (unsigned short*)(ws + 109051904);   //  4 MB
  unsigned short* Yb   = (unsigned short*)(ws + 113246208);   // 16 MB

  k_cvt_bf16  <<<8388608 / 256, 256, 0, stream>>>(x, x_bf, 8388608);
  k_cvt_bf16_T<<<6291456 / 256, 256, 0, stream>>>(W_attn, wa_t, 2048, 3072);
  k_cvt_bf16_T<<<4194304 / 256, 256, 0, stream>>>(W_proj, wp_t, 2048, 2048);

  // qkv = x @ W_attn   (M=4096, N=3072, K=2048)
  k_gemm_bf16<<<dim3(24, 32), 256, 0, stream>>>(x_bf, wa_t, qkv, 4096, 3072, 2048);

  // RoPE + permutation scatter -> Q/K/Vt bf16
  k_rope_scatter<<<6291456 / 256, 256, 0, stream>>>(qkv, freqs, Qb, Kb, Vb);

  // banded flash attention -> Yb bf16
  k_attn<<<dim3(256, 16), 32, 0, stream>>>(Qb, Kb, Vb, Yb);

  // out = y @ W_proj   (M=4096, N=2048, K=2048)
  k_gemm_bf16<<<dim3(16, 32), 256, 0, stream>>>(Yb, wp_t, out, 4096, 2048, 2048);
}
